// EdgeConv_7782480740941
// MI455X (gfx1250) — matrix operand order repaired and checked by reading, not compiled, not run on hardware
//
#include <hip/hip_runtime.h>
#include <hip/hip_bf16.h>

typedef __attribute__((ext_vector_type(16))) _Float16 v16h;
typedef __attribute__((ext_vector_type(8)))  _Float16 v8h;
typedef __attribute__((ext_vector_type(4)))  _Float16 h4;
typedef __attribute__((ext_vector_type(2)))  _Float16 h2;
typedef __attribute__((ext_vector_type(8)))  float    v8f;
typedef __attribute__((ext_vector_type(4)))  float    f4;

#define IN_FEAT   128
#define H_FEAT    64
#define OUT_FEAT  128

// LDS row strides in halfs (padded: 16B-aligned rows, dodge 64-bank conflicts)
#define W1_STRIDE 72    // [128][72]   W1 row-major (K x N), 144B rows
#define W2_STRIDE 136   // [64][136]   W2 row-major (K x N), 272B rows
#define HID_STRIDE 72   // per-wave [16][72]

// f32 pair -> packed f16 (v_cvt_pkrtz_f16_f32), one VGPR write
__device__ __forceinline__ h2 pk16(float a, float b) {
    return __builtin_bit_cast(h2, __builtin_amdgcn_cvt_pkrtz(a, b));
}

union AFrag { h2 p[8]; v16h v; };
union H4u   { h2 p[2]; h4 v; };

__global__ __launch_bounds__(256)
void edgeconv_wmma_kernel(const float* __restrict__ h,
                          const float* __restrict__ he,
                          const int*   __restrict__ src,
                          const int*   __restrict__ dst,
                          const float* __restrict__ W1,
                          const float* __restrict__ b1,
                          const float* __restrict__ W2,
                          const float* __restrict__ b2,
                          float* __restrict__ out,
                          int nEdges)
{
    __shared__ __align__(16) _Float16 sW1[IN_FEAT * W1_STRIDE];          // 18432 B
    __shared__ __align__(16) _Float16 sW2[H_FEAT  * W2_STRIDE];          // 17408 B
    __shared__ __align__(16) _Float16 sHid[8 * 16 * HID_STRIDE];         // 18432 B

    const int tid = threadIdx.x;

    // ---- Stage weights into LDS as f16 (vectorized: f4 loads, b64 DS stores) ----
    {   // W1 [128][64]: 2 threads/row, 32 contiguous cols each
        const int k  = tid >> 1;
        const int n0 = (tid & 1) << 5;
        const float*    gp = W1  + k * H_FEAT   + n0;
        _Float16*       lp = sW1 + k * W1_STRIDE + n0;
#pragma unroll
        for (int j = 0; j < 8; ++j) {
            f4 w = *(const f4*)(gp + 4 * j);             // 16B global load
            H4u t; t.p[0] = pk16(w.x, w.y); t.p[1] = pk16(w.z, w.w);
            *(h4*)(lp + 4 * j) = t.v;                    // 8B DS store
        }
    }
    {   // W2 [64][128]: 4 threads/row, 32 contiguous cols each
        const int k  = tid >> 2;
        const int n0 = (tid & 3) << 5;
        const float*    gp = W2  + k * OUT_FEAT  + n0;
        _Float16*       lp = sW2 + k * W2_STRIDE + n0;
#pragma unroll
        for (int j = 0; j < 8; ++j) {
            f4 w = *(const f4*)(gp + 4 * j);
            H4u t; t.p[0] = pk16(w.x, w.y); t.p[1] = pk16(w.z, w.w);
            *(h4*)(lp + 4 * j) = t.v;
        }
    }
    __syncthreads();

    const int wave = tid >> 5;
    const int lane = tid & 31;
    const int mrow = lane & 15;      // A row / C column / B n-index
    const int hsel = lane >> 4;      // lane-half select

    const int tile = blockIdx.x * 8 + wave;
    const int base = tile * 16;
    if (base >= nEdges) return;      // only barrier is above; safe
    const bool fullTile = (base + 16 <= nEdges);

    // Early prefetch of this tile's he rows (streamed later in the epilogue)
    {
        const float* hp = he + (long long)(base + mrow) * OUT_FEAT + 64 * hsel;
        __builtin_prefetch(hp, 0, 0);
        __builtin_prefetch(hp + 32, 0, 0);
    }

    // ---- Build A1 (16 edges x 128, f16): b128 gathers + pkrtz packing ----
    int e = base + mrow;
    if (e >= nEdges) e = nEdges - 1;                     // clamp (E%16==0 normally)
    const float* __restrict__ hs = h + (long long)src[e] * IN_FEAT;
    const float* __restrict__ hd = h + (long long)dst[e] * IN_FEAT;

    AFrag A1[4];
#pragma unroll
    for (int kf = 0; kf < 4; ++kf) {
        const float* ps = hs + kf * 32 + 8 * hsel;       // 16B aligned
        const float* pd = hd + kf * 32 + 8 * hsel;
        f4 s0 = *(const f4*)(ps);      f4 d0 = *(const f4*)(pd);
        f4 s1 = *(const f4*)(ps + 4);  f4 d1 = *(const f4*)(pd + 4);
        f4 s2 = *(const f4*)(ps + 16); f4 d2 = *(const f4*)(pd + 16);
        f4 s3 = *(const f4*)(ps + 20); f4 d3 = *(const f4*)(pd + 20);
        A1[kf].p[0] = pk16(fabsf(d0.x - s0.x), fabsf(d0.y - s0.y));  // k=0,1
        A1[kf].p[1] = pk16(fabsf(d0.z - s0.z), fabsf(d0.w - s0.w));  // k=2,3
        A1[kf].p[2] = pk16(fabsf(d1.x - s1.x), fabsf(d1.y - s1.y));  // k=4,5
        A1[kf].p[3] = pk16(fabsf(d1.z - s1.z), fabsf(d1.w - s1.w));  // k=6,7
        A1[kf].p[4] = pk16(fabsf(d2.x - s2.x), fabsf(d2.y - s2.y));  // k=16,17
        A1[kf].p[5] = pk16(fabsf(d2.z - s2.z), fabsf(d2.w - s2.w));  // k=18,19
        A1[kf].p[6] = pk16(fabsf(d3.x - s3.x), fabsf(d3.y - s3.y));  // k=20,21
        A1[kf].p[7] = pk16(fabsf(d3.z - s3.z), fabsf(d3.w - s3.w));  // k=22,23
    }

    // ---- Layer 1: hid = relu(diff @ W1 + b1), spill as f16 to per-wave LDS ----
    _Float16* __restrict__ myHid = sHid + wave * 16 * HID_STRIDE;
    const int bk = mrow + 16 * hsel;                      // unused; B lane selects column n = mrow
#pragma unroll
    for (int nt = 0; nt < 4; ++nt) {                      // H_FEAT = 4 x 16
        v8f c = {};
#pragma unroll
        for (int kf = 0; kf < 4; ++kf) {                  // K = 4 x 32
            const _Float16* bp = sW1 + (kf * 32 + 8 * hsel) * W1_STRIDE + nt * 16 + mrow;
            v16h bf;
#pragma unroll
            for (int s = 0; s < 16; ++s)
                bf[s] = bp[((s & 7) + ((s >> 3) << 4)) * W1_STRIDE];
            c = __builtin_amdgcn_wmma_f32_16x16x32_f16(
                    false, A1[kf].v, false, bf, (short)0, c, false, false);
        }
        float bias = b1[nt * 16 + mrow];
#pragma unroll
        for (int r = 0; r < 8; ++r) {                     // C: row r+8*hsel, col mrow
            float x = fmaxf(c[r] + bias, 0.0f);
            myHid[(r + 8 * hsel) * HID_STRIDE + nt * 16 + mrow] = (_Float16)x;
        }
    }

    // ---- Rebuild hid as A2 fragments: contiguous b128 DS loads + shuffle ----
    // (same-wave LDS store->load; DS ops are in-order per wave)
    AFrag A2[2];
#pragma unroll
    for (int kf = 0; kf < 2; ++kf) {                      // K = 64 = 2 x 32
        const _Float16* ap = myHid + mrow * HID_STRIDE + kf * 32 + 8 * hsel;
        v8h lo = *(const v8h*)(ap);                       // k = 0..7   (+8h)
        v8h hi = *(const v8h*)(ap + 16);                  // k = 16..23 (+8h)
        A2[kf].v = __builtin_shufflevector(lo, hi,
                        0,1,2,3,4,5,6,7,8,9,10,11,12,13,14,15);
    }

    // ---- Layer 2 + residual + sigmoid (nontemporal stream of he/out) ----
    const int p0 = (base + 8 * hsel) * OUT_FEAT + mrow;   // row base for this lane
#pragma unroll
    for (int nt = 0; nt < 8; ++nt) {                      // OUT_FEAT = 8 x 16
        v8f c = {};
#pragma unroll
        for (int kf = 0; kf < 2; ++kf) {
            const _Float16* bp = sW2 + (kf * 32 + 8 * hsel) * W2_STRIDE + nt * 16 + mrow;
            v16h bf;
#pragma unroll
            for (int s = 0; s < 16; ++s)
                bf[s] = bp[((s & 7) + ((s >> 3) << 4)) * W2_STRIDE];
            c = __builtin_amdgcn_wmma_f32_16x16x32_f16(
                    false, A2[kf].v, false, bf, (short)0, c, false, false);
        }
        float bias = b2[nt * 16 + mrow];
        const float* __restrict__ hp  = he  + p0 + nt * 16;
        float*       __restrict__ op  = out + p0 + nt * 16;
        if (fullTile) {
#pragma unroll
            for (int r = 0; r < 8; ++r) {
                float x = c[r] + bias + __builtin_nontemporal_load(hp + r * OUT_FEAT);
                float sg = __builtin_amdgcn_rcpf(
                    1.0f + __builtin_amdgcn_exp2f(x * -1.44269504f));
                __builtin_nontemporal_store(sg, op + r * OUT_FEAT);
            }
        } else {
#pragma unroll
            for (int r = 0; r < 8; ++r) {
                if (base + r + 8 * hsel < nEdges) {
                    float x = c[r] + bias + __builtin_nontemporal_load(hp + r * OUT_FEAT);
                    float sg = __builtin_amdgcn_rcpf(
                        1.0f + __builtin_amdgcn_exp2f(x * -1.44269504f));
                    __builtin_nontemporal_store(sg, op + r * OUT_FEAT);
                }
            }
        }
    }
}

extern "C" void kernel_launch(void* const* d_in, const int* in_sizes, int n_in,
                              void* d_out, int out_size, void* d_ws, size_t ws_size,
                              hipStream_t stream) {
    (void)n_in; (void)out_size; (void)d_ws; (void)ws_size;
    const float* h   = (const float*)d_in[0];
    const float* he  = (const float*)d_in[1];
    const int*   src = (const int*)d_in[2];
    const int*   dst = (const int*)d_in[3];
    const float* W1  = (const float*)d_in[4];
    const float* b1  = (const float*)d_in[5];
    const float* W2  = (const float*)d_in[6];
    const float* b2  = (const float*)d_in[7];
    float* out = (float*)d_out;

    const int E = in_sizes[2];                 // 640000 edges
    const int nTiles = (E + 15) / 16;          // 16-edge tiles, one per wave
    const int blocks = (nTiles + 7) / 8;       // 8 waves per 256-thread block

    edgeconv_wmma_kernel<<<blocks, 256, 0, stream>>>(
        h, he, src, dst, W1, b1, W2, b2, out, E);
}
